// TransformerLayer_29712583753916
// MI455X (gfx1250) — matrix operand order repaired and checked by reading, not compiled, not run on hardware
//
#include <hip/hip_runtime.h>
#include <hip/hip_bf16.h>

typedef _Float16 f16;
typedef __attribute__((ext_vector_type(16))) _Float16 v16h;
typedef __attribute__((ext_vector_type(8)))  _Float16 v8h;
typedef __attribute__((ext_vector_type(8)))  float    v8f;
typedef __attribute__((ext_vector_type(4)))  unsigned int v4u;
typedef __attribute__((ext_vector_type(8)))  int      v8i;
typedef __attribute__((ext_vector_type(4)))  int      v4i;

constexpr int B_  = 2;
constexpr int T_  = 1024;
constexpr int D_  = 1024;
constexpr int H_  = 16;
constexpr int HD_ = 64;
constexpr int E_  = 8;
constexpr int NTOK = B_ * T_;                 // 2048
constexpr long DD  = (long)D_ * D_;           // 1M

#if defined(__has_builtin)
#if __has_builtin(__builtin_amdgcn_tensor_load_to_lds)
#define HAVE_TDM 1
#endif
#endif
#ifndef HAVE_TDM
#define HAVE_TDM 0
#endif

// ---------------------------------------------------------------------------
// Fragment helpers (wave32; A/B in this code's contiguous k order, C/D per CDNA5 ISA)
// A 16x32 f16 frag, element j: row = lane&15, k = half*16 + j
//   -> two contiguous 16B chunks per lane at [half*16] and [half*16+8]
// B 32x16 f16 frag, element j: n = lane&15, k = half*16 + j
//   -> one contiguous 32B chunk per lane at [half*16]
// C/D f32 16x16: element r: m = r + 8*half, n = lane&15
// ---------------------------------------------------------------------------
__device__ __forceinline__ v16h frag_from2(const f16* p0, const f16* p1) {
  v8h lo = *(const v8h*)p0;
  v8h hi = *(const v8h*)p1;
  return __builtin_shufflevector(lo, hi, 0,1,2,3,4,5,6,7,8,9,10,11,12,13,14,15);
}

#if HAVE_TDM
// ---------------------------------------------------------------------------
// Tensor Data Mover descriptor builders (CDNA5 ISA)
// 2-D tile: tile_dim0 elements per row (contraction dim, contiguous in both
// memory and LDS), tile_dim1 rows, row stride = tensor_dim0_stride elements.
// data_size = 1 (2-byte f16).
// ---------------------------------------------------------------------------
__device__ __forceinline__ v4u tdm_g0(unsigned lds_off, unsigned long long ga) {
  v4u g0;
  g0[0] = 1u;                                    // count=1, user mode, no gather
  g0[1] = lds_off;                               // lds_addr (bytes)
  g0[2] = (unsigned)(ga & 0xFFFFFFFFu);          // global_addr[31:0]
  g0[3] = (unsigned)((ga >> 32) & 0x1FFFFFFu)    // global_addr[56:32]
        | (2u << 30);                            // type = 2 ("image")
  return g0;
}
__device__ __forceinline__ v8i tdm_g1(unsigned tile_d0, unsigned tile_d1,
                                      unsigned long long stride) {
  const unsigned td0 = 0x7FFFFFFFu, td1 = 0x7FFFFFFFu;  // no OOB clipping
  v8i g1;
  g1[0] = (int)(1u << 16);                                   // data_size=2B
  g1[1] = (int)((td0 & 0xFFFFu) << 16);                      // tensor_dim0 lo
  g1[2] = (int)(((td0 >> 16) & 0xFFFFu) | ((td1 & 0xFFFFu) << 16));
  g1[3] = (int)(((td1 >> 16) & 0xFFFFu) | ((tile_d0 & 0xFFFFu) << 16));
  g1[4] = (int)(tile_d1 & 0xFFFFu);                          // tile_dim2 = 0
  g1[5] = (int)(unsigned)(stride & 0xFFFFFFFFu);             // dim0 stride lo
  g1[6] = (int)(unsigned)((stride >> 32) & 0xFFFFu);         // dim0 stride hi
  g1[7] = 0;                                                 // dim1 stride (unused, 2D)
  return g1;
}
__device__ __forceinline__ void tdm_load(v4u g0, v8i g1) {
  v4i z4 = {0, 0, 0, 0};
#if __clang_major__ >= 23
  v8i z8 = {0, 0, 0, 0, 0, 0, 0, 0};
  __builtin_amdgcn_tensor_load_to_lds(g0, g1, z4, z4, z8, 0);
#else
  __builtin_amdgcn_tensor_load_to_lds(g0, g1, z4, z4, 0);
#endif
}
#endif  // HAVE_TDM

// ---------------------------------------------------------------------------
// fp32 [K,N] (optionally E mats) -> f16 transposed [N,K]
// ---------------------------------------------------------------------------
__global__ void convt_kernel(const float* __restrict__ in, f16* __restrict__ out,
                             int K, int N, long total) {
  long KN = (long)K * N;
  for (long idx = blockIdx.x * (long)blockDim.x + threadIdx.x; idx < total;
       idx += (long)gridDim.x * blockDim.x) {
    long e   = idx / KN;
    long rem = idx - e * KN;
    long n   = rem / K;
    long k   = rem - n * K;
    out[idx] = (f16)in[e * KN + k * (long)N + n];
  }
}

// ---------------------------------------------------------------------------
// LayerNorm over D=1024; one block (256 thr) per row.
// ---------------------------------------------------------------------------
__global__ __launch_bounds__(256) void ln_kernel(
    const float* __restrict__ x, const float* __restrict__ g,
    const float* __restrict__ b, f16* __restrict__ out_h,
    float* __restrict__ out_f, float* __restrict__ pass) {
  __shared__ float red[256];
  const int row = blockIdx.x, tid = threadIdx.x;
  const float* xr = x + (size_t)row * D_;
  float xv[4];
  float s = 0.f;
  for (int i = 0; i < 4; i++) { xv[i] = xr[tid + i * 256]; s += xv[i]; }
  red[tid] = s; __syncthreads();
  for (int off = 128; off; off >>= 1) {
    if (tid < off) red[tid] += red[tid + off];
    __syncthreads();
  }
  float mean = red[0] * (1.f / D_);
  __syncthreads();
  float vs = 0.f;
  for (int i = 0; i < 4; i++) { float d = xv[i] - mean; vs += d * d; }
  red[tid] = vs; __syncthreads();
  for (int off = 128; off; off >>= 1) {
    if (tid < off) red[tid] += red[tid + off];
    __syncthreads();
  }
  float rstd = rsqrtf(red[0] * (1.f / D_) + 1e-5f);
  for (int i = 0; i < 4; i++) {
    int c = tid + i * 256;
    float y = (xv[i] - mean) * rstd * g[c] + b[c];
    size_t o = (size_t)row * D_ + c;
    out_h[o] = (f16)y;
    if (out_f) out_f[o] = y;
    if (pass)  pass[o]  = xv[i];
  }
}

// ---------------------------------------------------------------------------
// WMMA GEMM: C[M,N] = A[M,K](f16) * B[K,N] given as BT[N,K](f16).
// Double-buffered LDS; tiles DMA'd by the Tensor Data Mover (TENSORcnt),
// overlapped with WMMA on the other buffer. Epilogue fuses bias / exact GELU /
// per-row scale / residual / f16 requant.
// Block 256 = 8 waves; tile 128x128x32; wave = 64x32 = 4x2 WMMAs.
// ---------------------------------------------------------------------------
constexpr int BM = 128, BN = 128, BK = 32;
constexpr int TILE_H = BM * BK;          // halves per tile (BM == BN)

__global__ __launch_bounds__(256) void gemm_f16_kernel(
    const f16* __restrict__ A, const f16* __restrict__ BT,
    const float* __restrict__ bias,
    const float* __restrict__ rowscale, int rs_stride,
    const float* __restrict__ residual,
    float* __restrict__ out_f, int out_f_add,
    f16* __restrict__ out_h,
    int M, int N, int K, int act) {
  // single LDS arena so TDM descriptor offsets are static:
  // [A ping][A pong][B ping][B pong], 8KB each
  __shared__ __align__(128) f16 smem[4 * TILE_H];

  const int tid  = threadIdx.x;
  const int lane = tid & 31, wv = tid >> 5;
  const int half = lane >> 4, lrow = lane & 15;
  const int wm = wv >> 2, wn = wv & 3;       // waves: 2 (M) x 4 (N)
  const int m0 = blockIdx.y * BM, n0 = blockIdx.x * BN;

  v8f acc[4][2];
  for (int i = 0; i < 4; i++)
    for (int j = 0; j < 2; j++)
      for (int r = 0; r < 8; r++) acc[i][j][r] = 0.f;

  const int nk = K / BK;

#if HAVE_TDM
  const unsigned lds_base = (unsigned)(size_t)(void*)&smem[0];
  auto issue = [&](int step, int p) {
    unsigned long long ga = (unsigned long long)(size_t)A +
        ((size_t)m0 * K + (size_t)step * BK) * sizeof(f16);
    unsigned long long gb = (unsigned long long)(size_t)BT +
        ((size_t)n0 * K + (size_t)step * BK) * sizeof(f16);
    tdm_load(tdm_g0(lds_base + (unsigned)(p * TILE_H) * 2u, ga),
             tdm_g1(BK, BM, (unsigned)K));
    tdm_load(tdm_g0(lds_base + (unsigned)((2 + p) * TILE_H) * 2u, gb),
             tdm_g1(BK, BN, (unsigned)K));
  };
  if (wv == 0) issue(0, 0);
  for (int i = 0; i < nk; i++) {
    const int p = i & 1;
    const f16* sa = smem + p * TILE_H;        // runtime addrspacecast, no
    const f16* sb = smem + (2 + p) * TILE_H;  // static pointer initializers
    if (wv == 0) {
      if (i + 1 < nk) {
        issue(i + 1, p ^ 1);                         // prefetch next tile pair
        __builtin_amdgcn_s_wait_tensorcnt((short)2); // older pair complete
      } else {
        __builtin_amdgcn_s_wait_tensorcnt((short)0);
      }
    }
    __syncthreads();
    v16h af[4], bf[2];
    for (int mi = 0; mi < 4; mi++) {
      const f16* rp = sa + (wm * 64 + mi * 16 + lrow) * BK;
      af[mi] = frag_from2(rp + half * 16, rp + half * 16 + 8);
    }
    for (int ni = 0; ni < 2; ni++)
      bf[ni] = *(const v16h*)(sb + (wn * 32 + ni * 16 + lrow) * BK + half * 16);
    for (int mi = 0; mi < 4; mi++)
      for (int ni = 0; ni < 2; ni++)
        acc[mi][ni] = __builtin_amdgcn_wmma_f32_16x16x32_f16(
            false, af[mi], false, bf[ni], (short)0, acc[mi][ni], false, false);
    __syncthreads();
  }
#else
  for (int i = 0; i < nk; i++) {
    const int k0 = i * BK;
    f16* sa = smem;
    f16* sb = smem + 2 * TILE_H;
    for (int t = 0; t < 2; t++) {
      int idx = tid + t * 256;
      int row = idx >> 2, ch = (idx & 3) * 8;
      *(v8h*)(sa + row * BK + ch) =
          *(const v8h*)&A [(size_t)(m0 + row) * K + k0 + ch];
      *(v8h*)(sb + row * BK + ch) =
          *(const v8h*)&BT[(size_t)(n0 + row) * K + k0 + ch];
    }
    __syncthreads();
    v16h af[4], bf[2];
    for (int mi = 0; mi < 4; mi++) {
      const f16* rp = sa + (wm * 64 + mi * 16 + lrow) * BK;
      af[mi] = frag_from2(rp + half * 16, rp + half * 16 + 8);
    }
    for (int ni = 0; ni < 2; ni++)
      bf[ni] = *(const v16h*)(sb + (wn * 32 + ni * 16 + lrow) * BK + half * 16);
    for (int mi = 0; mi < 4; mi++)
      for (int ni = 0; ni < 2; ni++)
        acc[mi][ni] = __builtin_amdgcn_wmma_f32_16x16x32_f16(
            false, af[mi], false, bf[ni], (short)0, acc[mi][ni], false, false);
    __syncthreads();
  }
#endif

  for (int mi = 0; mi < 4; mi++)
    for (int ni = 0; ni < 2; ni++)
      for (int r = 0; r < 8; r++) {
        int gm = m0 + wm * 64 + mi * 16 + r + half * 8;
        int gn = n0 + wn * 32 + ni * 16 + lrow;
        float v = acc[mi][ni][r];
        if (bias) v += bias[gn];
        if (act == 1) v = 0.5f * v * (1.0f + erff(v * 0.70710678f));
        if (rowscale) v *= rowscale[(size_t)gm * rs_stride];
        if (residual) v += residual[(size_t)gm * N + gn];
        size_t o = (size_t)gm * N + gn;
        if (out_f) { if (out_f_add) out_f[o] += v; else out_f[o] = v; }
        if (out_h) out_h[o] = (f16)v;
      }
}

// ---------------------------------------------------------------------------
// Flash-style attention. Grid (T/64, B*H), 128 threads = 4 waves.
// Each wave owns 16 q-rows; K/V streamed in 64-key LDS blocks; WMMA for
// Q*K^T and P*V; online softmax via intra-half shfl reductions.
// ---------------------------------------------------------------------------
__global__ __launch_bounds__(128) void attn_kernel(
    const f16* __restrict__ Q, const f16* __restrict__ Kp,
    const f16* __restrict__ Vp, f16* __restrict__ O) {
  __shared__ __align__(64) f16 lQ [64][64];
  __shared__ __align__(64) f16 lK [64][64];
  __shared__ __align__(64) f16 lVt[64][64];    // [hd][key]
  __shared__ __align__(64) f16 lP [4][16][64]; // per-wave P staging

  const int bh = blockIdx.y, b = bh / H_, h = bh % H_;
  const int q0 = blockIdx.x * 64;
  const int tid  = threadIdx.x;
  const int lane = tid & 31, wv = tid >> 5;
  const int half = lane >> 4, lrow = lane & 15;

  for (int i = 0; i < 4; i++) {
    int idx = tid + i * 128;
    int row = idx >> 3, ch = (idx & 7) * 8;
    *(v8h*)&lQ[row][ch] =
        *(const v8h*)&Q[((size_t)(b * T_ + q0 + row)) * D_ + h * HD_ + ch];
  }
  __syncthreads();
  v16h aQ[2];
  {
    const f16* rp = &lQ[wv * 16 + lrow][0];
    for (int s = 0; s < 2; s++)
      aQ[s] = frag_from2(rp + s * 32 + half * 16, rp + s * 32 + half * 16 + 8);
  }

  float mrun[8], lrun[8];
  v8f oacc[4];
  for (int r = 0; r < 8; r++) { mrun[r] = -1e30f; lrun[r] = 0.f; }
  for (int nt = 0; nt < 4; nt++)
    for (int r = 0; r < 8; r++) oacc[nt][r] = 0.f;

  const float escale = 0.03125f;  // 1/sqrt(D), per reference

  for (int kb = 0; kb < T_ / 64; kb++) {
    __syncthreads();
    for (int i = 0; i < 4; i++) {
      int idx = tid + i * 128;
      int row = idx >> 3, ch = (idx & 7) * 8;
      size_t src = ((size_t)(b * T_ + kb * 64 + row)) * D_ + h * HD_ + ch;
      *(v8h*)&lK[row][ch] = *(const v8h*)&Kp[src];
      v8h vvv = *(const v8h*)&Vp[src];
      for (int j = 0; j < 8; j++) lVt[ch + j][row] = vvv[j];
    }
    __syncthreads();

    v8f sacc[4];
    for (int nt = 0; nt < 4; nt++)
      for (int r = 0; r < 8; r++) sacc[nt][r] = 0.f;
    for (int s = 0; s < 2; s++)
      for (int nt = 0; nt < 4; nt++) {
        v16h bK = *(const v16h*)&lK[nt * 16 + lrow][s * 32 + half * 16];
        sacc[nt] = __builtin_amdgcn_wmma_f32_16x16x32_f16(
            false, aQ[s], false, bK, (short)0, sacc[nt], false, false);
      }

    float rowsc[8];
    for (int r = 0; r < 8; r++) {
      float mx = -1e30f;
      for (int nt = 0; nt < 4; nt++) mx = fmaxf(mx, sacc[nt][r] * escale);
      for (int m = 8; m; m >>= 1) mx = fmaxf(mx, __shfl_xor(mx, m, 32));
      float mnew = fmaxf(mrun[r], mx);
      float sc = expf(mrun[r] - mnew);
      float ps = 0.f;
      int mrow = r + half * 8;
      for (int nt = 0; nt < 4; nt++) {
        float p = expf(sacc[nt][r] * escale - mnew);
        lP[wv][mrow][nt * 16 + lrow] = (f16)p;
        ps += p;
      }
      for (int m = 8; m; m >>= 1) ps += __shfl_xor(ps, m, 32);
      lrun[r] = lrun[r] * sc + ps;
      mrun[r] = mnew;
      rowsc[r] = sc;
    }
    for (int nt = 0; nt < 4; nt++)
      for (int r = 0; r < 8; r++) oacc[nt][r] *= rowsc[r];

    v16h aP[2];
    {
      const f16* rp = &lP[wv][lrow][0];
      for (int s = 0; s < 2; s++)
        aP[s] = frag_from2(rp + s * 32 + half * 16, rp + s * 32 + half * 16 + 8);
    }
    for (int s = 0; s < 2; s++)
      for (int nt = 0; nt < 4; nt++) {
        v16h bV = *(const v16h*)&lVt[nt * 16 + lrow][s * 32 + half * 16];
        oacc[nt] = __builtin_amdgcn_wmma_f32_16x16x32_f16(
            false, aP[s], false, bV, (short)0, oacc[nt], false, false);
      }
  }

  for (int nt = 0; nt < 4; nt++)
    for (int r = 0; r < 8; r++) {
      int mrow = r + half * 8;
      float v = oacc[nt][r] / lrun[r];
      O[((size_t)(b * T_ + q0 + wv * 16 + mrow)) * D_ + h * HD_ + nt * 16 + lrow] =
          (f16)v;
    }
}

// ---------------------------------------------------------------------------
// Gating: one wave per token. logits = xn @ Wg + bg; top-2; softmax -> dense
// combine[N,E].
// ---------------------------------------------------------------------------
__global__ __launch_bounds__(32) void gate_kernel(
    const float* __restrict__ xn, const float* __restrict__ Wg,
    const float* __restrict__ bg, float* __restrict__ combine) {
  const int row = blockIdx.x, lane = threadIdx.x;
  float acc[E_];
  for (int e = 0; e < E_; e++) acc[e] = 0.f;
  const float* xr = xn + (size_t)row * D_;
  for (int d = lane; d < D_; d += 32) {
    float xv = xr[d];
    for (int e = 0; e < E_; e++) acc[e] += xv * Wg[(size_t)d * E_ + e];
  }
  for (int e = 0; e < E_; e++)
    for (int m = 16; m; m >>= 1) acc[e] += __shfl_xor(acc[e], m, 32);
  if (lane == 0) {
    float lg[E_];
    for (int e = 0; e < E_; e++) lg[e] = acc[e] + bg[e];
    int i1 = 0;
    for (int e = 1; e < E_; e++) if (lg[e] > lg[i1]) i1 = e;
    int i2 = -1;
    for (int e = 0; e < E_; e++)
      if (e != i1 && (i2 < 0 || lg[e] > lg[i2])) i2 = e;
    float z  = expf(lg[i2] - lg[i1]);
    float w1 = 1.f / (1.f + z), w2 = z / (1.f + z);
    float* cr = combine + (size_t)row * E_;
    for (int e = 0; e < E_; e++) cr[e] = 0.f;
    cr[i1] = w1; cr[i2] = w2;
  }
}

// ---------------------------------------------------------------------------
extern "C" void kernel_launch(void* const* d_in, const int* in_sizes, int n_in,
                              void* d_out, int out_size, void* d_ws,
                              size_t ws_size, hipStream_t stream) {
  const float* key  = (const float*)d_in[0];
  const float* ln0g = (const float*)d_in[1];
  const float* ln0b = (const float*)d_in[2];
  const float* Wv   = (const float*)d_in[3];
  const float* Wk   = (const float*)d_in[4];
  const float* Wq   = (const float*)d_in[5];
  const float* Wo   = (const float*)d_in[6];
  const float* ln1g = (const float*)d_in[7];
  const float* ln1b = (const float*)d_in[8];
  const float* Wg   = (const float*)d_in[9];
  const float* bg   = (const float*)d_in[10];
  const float* W1   = (const float*)d_in[11];
  const float* b1   = (const float*)d_in[12];
  const float* W2   = (const float*)d_in[13];
  const float* b2   = (const float*)d_in[14];
  float* out = (float*)d_out;

  char* wsb = (char*)d_ws;
  size_t off = 0;
  auto alloc = [&](size_t bytes) {
    void* p = wsb + off;
    off += (bytes + 255) & ~(size_t)255;
    return p;
  };
  f16*  Wv_h  = (f16*)alloc(DD * 2);
  f16*  Wk_h  = (f16*)alloc(DD * 2);
  f16*  Wq_h  = (f16*)alloc(DD * 2);
  f16*  Wo_h  = (f16*)alloc(DD * 2);
  f16*  W1_h  = (f16*)alloc((size_t)E_ * DD * 2);
  f16*  W2_h  = (f16*)alloc((size_t)E_ * DD * 2);
  f16*  kn_h  = (f16*)alloc((size_t)NTOK * D_ * 2);
  f16*  q_h   = (f16*)alloc((size_t)NTOK * D_ * 2);
  f16*  k_h   = (f16*)alloc((size_t)NTOK * D_ * 2);
  f16*  v_h   = (f16*)alloc((size_t)NTOK * D_ * 2);
  f16*  o_h   = (f16*)alloc((size_t)NTOK * D_ * 2);
  f16*  xn_h  = (f16*)alloc((size_t)NTOK * D_ * 2);
  f16*  h_h   = (f16*)alloc((size_t)NTOK * D_ * 2);
  float* a_f  = (float*)alloc((size_t)NTOK * D_ * 4);
  float* xn_f = (float*)alloc((size_t)NTOK * D_ * 4);
  float* comb = (float*)alloc((size_t)NTOK * E_ * 4);

  // 1) weights -> f16, transposed [N,K]
  convt_kernel<<<2048, 256, 0, stream>>>(Wv, Wv_h, D_, D_, DD);
  convt_kernel<<<2048, 256, 0, stream>>>(Wk, Wk_h, D_, D_, DD);
  convt_kernel<<<2048, 256, 0, stream>>>(Wq, Wq_h, D_, D_, DD);
  convt_kernel<<<2048, 256, 0, stream>>>(Wo, Wo_h, D_, D_, DD);
  convt_kernel<<<4096, 256, 0, stream>>>(W1, W1_h, D_, D_, (long)E_ * DD);
  convt_kernel<<<4096, 256, 0, stream>>>(W2, W2_h, D_, D_, (long)E_ * DD);

  // 2) LN0 -> kn (f16)
  ln_kernel<<<NTOK, 256, 0, stream>>>(key, ln0g, ln0b, kn_h, nullptr, nullptr);

  dim3 ggrid(D_ / BN, NTOK / BM);
  // 3) QKV projections
  gemm_f16_kernel<<<ggrid, 256, 0, stream>>>(kn_h, Wq_h, nullptr, nullptr, 0,
      nullptr, nullptr, 0, q_h, NTOK, D_, D_, 0);
  gemm_f16_kernel<<<ggrid, 256, 0, stream>>>(kn_h, Wk_h, nullptr, nullptr, 0,
      nullptr, nullptr, 0, k_h, NTOK, D_, D_, 0);
  gemm_f16_kernel<<<ggrid, 256, 0, stream>>>(kn_h, Wv_h, nullptr, nullptr, 0,
      nullptr, nullptr, 0, v_h, NTOK, D_, D_, 0);

  // 4) attention
  attn_kernel<<<dim3(T_ / 64, B_ * H_), 128, 0, stream>>>(q_h, k_h, v_h, o_h);

  // 5) output projection + residual (a = o@Wo + key)
  gemm_f16_kernel<<<ggrid, 256, 0, stream>>>(o_h, Wo_h, nullptr, nullptr, 0,
      key, a_f, 0, nullptr, NTOK, D_, D_, 0);

  // 6) LN1 -> xn (f16+f32), and copy a into d_out (residual base)
  ln_kernel<<<NTOK, 256, 0, stream>>>(a_f, ln1g, ln1b, xn_h, xn_f, out);

  // 7) gating -> dense combine[N,E]
  gate_kernel<<<NTOK, 32, 0, stream>>>(xn_f, Wg, bg, comb);

  // 8) MoE: per expert, h = gelu(xn@W1+b1); out += combine[:,e]*(h@W2+b2)
  for (int e = 0; e < E_; e++) {
    gemm_f16_kernel<<<ggrid, 256, 0, stream>>>(xn_h, W1_h + (size_t)e * DD,
        b1 + (size_t)e * D_, nullptr, 0, nullptr, nullptr, 0, h_h,
        NTOK, D_, D_, 1);
    gemm_f16_kernel<<<ggrid, 256, 0, stream>>>(h_h, W2_h + (size_t)e * DD,
        b2 + (size_t)e * D_, comb + e, E_, nullptr, out, 1, nullptr,
        NTOK, D_, D_, 0);
  }
}
